// AP_NN_89249420411447
// MI455X (gfx1250) — matrix operand order repaired and checked by reading, not compiled, not run on hardware
//
#include <hip/hip_runtime.h>
#include <stdint.h>

#define T_NO   501
#define NBASIS 30
#define BATCH  64
#define TT     2048
#define WGB    16      // batches per workgroup
#define BLK    16      // timesteps per WMMA block
#define KTILES 16      // 512-wide history window (16 x K32 tiles)
#define RING   1024    // LDS spike ring width (power of 2 >= 512+16)

typedef __attribute__((ext_vector_type(16))) _Float16 v16h;
typedef __attribute__((ext_vector_type(8)))  _Float16 v8h;
typedef __attribute__((ext_vector_type(8)))  float    v8f;

// ---------------- Threefry-2x32 (20 rounds), matches jax.random ----------------
__device__ __forceinline__ void tf2x32(uint32_t k0, uint32_t k1,
                                       uint32_t x0, uint32_t x1,
                                       uint32_t& o0, uint32_t& o1) {
  uint32_t ks0 = k0, ks1 = k1, ks2 = k0 ^ k1 ^ 0x1BD11BDAu;
  x0 += ks0; x1 += ks1;
#define RR(r) { x0 += x1; x1 = (x1 << (r)) | (x1 >> (32 - (r))); x1 ^= x0; }
  RR(13) RR(15) RR(26) RR(6)   x0 += ks1; x1 += ks2 + 1u;
  RR(17) RR(29) RR(16) RR(24)  x0 += ks2; x1 += ks0 + 2u;
  RR(13) RR(15) RR(26) RR(6)   x0 += ks0; x1 += ks1 + 3u;
  RR(17) RR(29) RR(16) RR(24)  x0 += ks1; x1 += ks2 + 4u;
  RR(13) RR(15) RR(26) RR(6)   x0 += ks2; x1 += ks0 + 5u;
#undef RR
  o0 = x0; o1 = x1;
}

// ---------------- K1: raised-cosine basis -> flipped refractory kernel --------
__global__ void ap_kern_build(const float* __restrict__ Wref, float* __restrict__ kernf) {
  int t = threadIdx.x;               // kern[j] = sum_i basis[i, 500-j] * W[i]
  if (t >= 512) return;
  double val = 0.0;
  if (t <= 500) {
    const double PI = 3.14159265358979323846;
    double tt  = (double)(500 - t);
    double raw = 7.5 * log(tt + 1.0 + 1e-7);
    for (int i = 0; i < NBASIS; ++i) {
      double phi = 1.5707963267948966 * (double)i;
      double d = raw - phi;
      if (d >= -PI && d <= PI)
        val += (double)Wref[i] * (0.5 * cos(d) + 0.5);
    }
  }
  kernf[t] = (float)val;
}

// ---------------- K2: sequential jax.random.split chain, key(42) --------------
__global__ void ap_key_chain(uint32_t* __restrict__ subkeys) {
  if (threadIdx.x != 0 || blockIdx.x != 0) return;
  uint32_t k0 = 0u, k1 = 42u;              // jax.random.key(42) -> (0, 42)
  for (int t = 0; t < TT; ++t) {
    uint32_t a0, b0, a1, b1;
    tf2x32(k0, k1, 0u, 2u, a0, b0);        // counts [0,1,2,3] -> pairs (0,2),(1,3)
    tf2x32(k0, k1, 1u, 3u, a1, b1);
    k0 = a0; k1 = a1;                      // new key = row0, sub = row1
    subkeys[2*t] = b0; subkeys[2*t+1] = b1;
  }
}

// ---------------- K3: pointwise MLP nn_out = w2.tanh(w1.[V;D]+b1)+b2 ----------
__global__ void ap_nn_mlp(const float* __restrict__ V, const float* __restrict__ D,
                          const float* __restrict__ w1, const float* __restrict__ b1,
                          const float* __restrict__ w2, const float* __restrict__ b2,
                          float* __restrict__ nn) {
  int idx = blockIdx.x * blockDim.x + threadIdx.x;
  if (idx >= BATCH * TT) return;
  float v = V[idx], d = D[idx];
  float acc = b2[0];
#pragma unroll
  for (int k = 0; k < 5; ++k) {
    float h = tanhf(w1[2*k] * v + w1[2*k+1] * d + b1[k]);
    acc += w2[k] * h;
  }
  nn[idx] = acc;
}

// ---------------- K4: blocked autoregressive scan (WMMA history + tail) -------
__global__ __launch_bounds__(128, 1)
void ap_scan(const float* __restrict__ kernf, const uint32_t* __restrict__ subkeys,
             const float* __restrict__ nn, float* __restrict__ S_out,
             float* __restrict__ P_out) {
  __shared__ alignas(32) _Float16 S_lds[WGB * RING];          // 32 KB spike ring (abs time mod 1024)
  __shared__ alignas(32) _Float16 B_lds[KTILES * 32 * 16];    // 16 KB block-invariant B fragments
  __shared__ float H_lds[4 * WGB * BLK];                      //  4 KB per-wave partial sums

  const int tid = threadIdx.x;
  const int wv  = tid >> 5;
  const int L   = tid & 31;
  const int b0g = blockIdx.x * WGB;    // this WG owns batches [b0g, b0g+16)

  // zero the ring
  for (int idx = tid; idx < WGB * RING / 2; idx += 128)
    ((uint32_t*)S_lds)[idx] = 0u;

  // Build B fragments: W[k,i] = kern[k+5-i] for valid range, window base = t0-496.
  // B fragment layout used here (contiguous k, as in A): N = lane&15, K = 32*kt + 16*(lane>>4) + h.
  for (int idx = tid; idx < KTILES * 32 * 16; idx += 128) {
    int kt = idx >> 9, rem = idx & 511, lane = rem >> 4, h = rem & 15;
    int n = lane & 15;
    int k = 32 * kt + 16 * (lane >> 4) + h;
    int j = k + 5 - n;
    float wvv = (j >= 0 && j <= 500) ? kernf[j] : 0.0f;
    B_lds[idx] = (_Float16)wvv;
  }

  // per-lane constants for the sequential tail
  float kw[16];    // kw[d] = kern[501-d]  (in-block feedback, d = 1..15)
  float kc[5];     // kc[m] = kern[m]      (pre-window correction)
  kw[0] = 0.0f;
#pragma unroll
  for (int d = 1; d <= 15; ++d) kw[d] = kernf[T_NO - d];
#pragma unroll
  for (int m = 0; m < 5; ++m) kc[m] = kernf[m];

  __syncthreads();

  const int m16 = L & 15;
  const int kb8 = 8 * (L >> 4);

  for (int blk = 0; blk < TT / BLK; ++blk) {
    const int t0 = blk * BLK;

    // ---- Phase A: H[b,i] = sum_{k<512} ring[b, t0-496+k] * W[k,i], K-split 4 tiles/wave
    v8f acc = {};
#pragma unroll
    for (int j = 0; j < 4; ++j) {
      int kt = 4 * wv + j;
      v16h bfrag = *(const v16h*)&B_lds[kt * 512 + L * 16];
      int s1 = t0 - 496 + 32 * kt + 2 * kb8;       // multiple of 16: no wrap inside a run
      v8h alo = *(const v8h*)&S_lds[m16 * RING + (s1 & (RING - 1))];
      v8h ahi = *(const v8h*)&S_lds[m16 * RING + ((s1 + 8) & (RING - 1))];
      v16h afrag = __builtin_shufflevector(alo, ahi,
          0,1,2,3,4,5,6,7,8,9,10,11,12,13,14,15);
      acc = __builtin_amdgcn_wmma_f32_16x16x32_f16(
          false, afrag, false, bfrag, (short)0, acc, false, false);
    }
#pragma unroll
    for (int r = 0; r < 8; ++r)                    // D: M = r + 8*(L>>4), N = L&15
      H_lds[wv * 256 + (r + kb8) * 16 + m16] = acc[r];

    __syncthreads();

    // ---- Phase B: 16-step sequential tail, one batch per lane (lanes 0..15)
    if (tid < 16) {
      const int l = tid;
      const int b = b0g + l;
      const uint32_t q = (uint32_t)(b & 31);
      float nnarr[16];
      const float4* nnp = (const float4*)(nn + b * TT + t0);
#pragma unroll
      for (int v = 0; v < 4; ++v) {
        float4 f = nnp[v];
        nnarr[4*v+0] = f.x; nnarr[4*v+1] = f.y; nnarr[4*v+2] = f.z; nnarr[4*v+3] = f.w;
      }
      float spk[16];
#pragma unroll
      for (int i = 0; i < 16; ++i) {
        const int t = t0 + i;
        float x = nnarr[i]
                + H_lds[0*256 + l*16 + i] + H_lds[1*256 + l*16 + i]
                + H_lds[2*256 + l*16 + i] + H_lds[3*256 + l*16 + i];
        if (i <= 4) {                               // taps older than the 512 window
#pragma unroll
          for (int mm = 0; mm + i <= 4; ++mm)
            x += (float)S_lds[l * RING + ((t - T_NO + mm) & (RING - 1))] * kc[mm];
        }
#pragma unroll
        for (int jj = 0; jj < i; ++jj)              // in-block autoregressive feedback
          x += spk[jj] * kw[i - jj];
        float P = 1.0f / (1.0f + __expf(-x));
        uint32_t o0, o1;                            // bernoulli bits: pair (q, q+32)
        tf2x32(subkeys[2*t], subkeys[2*t+1], q, q + 32u, o0, o1);
        uint32_t bits = (b < 32) ? o0 : o1;
        float u = __uint_as_float((bits >> 9) | 0x3f800000u) - 1.0f;
        float s = (u < P) ? 1.0f : 0.0f;
        spk[i] = s;
        S_out[b * TT + t] = s;
        P_out[b * TT + t] = P;
        S_lds[l * RING + (t & (RING - 1))] = (_Float16)s;
      }
    } else if (tid >= 32 && tid < 48) {
      // wave 1, lanes 0..15: clear next block's ring slots (expired spikes from t0-1008..)
      int r = tid - 32;
      int base = r * RING + ((t0 + 16) & (RING - 1));   // 32B aligned, no wrap in 16
      uint4 z = make_uint4(0u, 0u, 0u, 0u);
      *(uint4*)&S_lds[base]     = z;
      *(uint4*)&S_lds[base + 8] = z;
    }
    __syncthreads();
  }
}

// ------------------------------- launcher -------------------------------------
extern "C" void kernel_launch(void* const* d_in, const int* in_sizes, int n_in,
                              void* d_out, int out_size, void* d_ws, size_t ws_size,
                              hipStream_t stream) {
  const float* V    = (const float*)d_in[0];
  const float* D    = (const float*)d_in[1];
  const float* w1   = (const float*)d_in[2];
  const float* b1   = (const float*)d_in[3];
  const float* w2   = (const float*)d_in[4];
  const float* b2   = (const float*)d_in[5];
  const float* Wref = (const float*)d_in[6];

  float* out   = (float*)d_out;
  float* S_out = out;                     // [64,2048]
  float* P_out = out + BATCH * TT;        // [64,2048]

  // workspace layout: kern f32[512] | subkeys u32[2*2048] | nn f32[64*2048]
  float*    kernf   = (float*)d_ws;
  uint32_t* subkeys = (uint32_t*)((char*)d_ws + 2048);
  float*    nn      = (float*)((char*)d_ws + 2048 + 4 * 2 * TT);

  hipLaunchKernelGGL(ap_kern_build, dim3(1),   dim3(512), 0, stream, Wref, kernf);
  hipLaunchKernelGGL(ap_key_chain,  dim3(1),   dim3(32),  0, stream, subkeys);
  hipLaunchKernelGGL(ap_nn_mlp,     dim3((BATCH*TT + 255)/256), dim3(256), 0, stream,
                     V, D, w1, b1, w2, b2, nn);
  hipLaunchKernelGGL(ap_scan,       dim3(BATCH / WGB), dim3(128), 0, stream,
                     kernf, subkeys, nn, S_out, P_out);
}
